// MultiHeadAttention_90074054132340
// MI455X (gfx1250) — matrix operand order repaired and checked by reading, not compiled, not run on hardware
//
#include <hip/hip_runtime.h>

typedef __attribute__((ext_vector_type(16))) _Float16 v16h;
typedef __attribute__((ext_vector_type(8)))  _Float16 v8h;
typedef __attribute__((ext_vector_type(4)))  _Float16 v4h;
typedef __attribute__((ext_vector_type(8)))  float    v8f;
typedef __attribute__((ext_vector_type(4)))  unsigned int u32x4;
typedef __attribute__((ext_vector_type(8)))  int      i32x8;
typedef __attribute__((ext_vector_type(4)))  int      i32x4;

union H16 { v16h v; v8h h[2]; _Float16 e[16]; };
union F8  { v8f  v; float f[8]; };

constexpr int D_MODEL = 1024;
constexpr int S_LEN   = 2048;
constexpr int BATCH   = 4;
constexpr int NHEAD   = 16;
constexpr int DKH     = 64;
constexpr int MTOT    = BATCH * S_LEN;   // 8192

#if __has_builtin(__builtin_amdgcn_tensor_load_to_lds) && __has_builtin(__builtin_amdgcn_s_wait_tensorcnt)
#define USE_TDM 1
#else
#define USE_TDM 0
#endif

#if USE_TDM
// LDS byte offset: generic pointers to LDS keep the workgroup-relative offset
// in addr[31:0] (ISA 10.2 aperture mapping), so truncation recovers it.
__device__ __forceinline__ unsigned int lds_off(const void* p) {
  return (unsigned int)(size_t)p;
}

// 2-D f16 tile load via Tensor Data Mover (D# groups per ISA 8.3/8.4).
// tile_x halves per row, tile_y rows, row stride in halves; LDS rows padded:
// pad after 2^(ic+1) dwords by (pa+1) dwords.
__device__ __forceinline__ void tdm_load_2d(const void* gsrc, unsigned int ldsOff,
                                            unsigned int tile_x, unsigned int tile_y,
                                            unsigned long long stride_x,
                                            unsigned int ic, unsigned int pa) {
  unsigned long long ga = (unsigned long long)(size_t)gsrc;
  u32x4 g0;
  g0[0] = 1u;                                          // count=1, user mode
  g0[1] = ldsOff;                                      // lds_addr
  g0[2] = (unsigned int)ga;                            // global_addr[31:0]
  g0[3] = (unsigned int)((ga >> 32) & 0x01FFFFFFu) | (2u << 30);  // addr[56:32], type=2
  const unsigned int dim0 = 0x40000000u, dim1 = 0x40000000u;      // no OOB clip
  i32x8 g1;
  g1[0] = (int)((1u << 16) | (1u << 20) | (ic << 22) | (pa << 25)); // 2B elems, pad
  g1[1] = (int)((dim0 & 0xFFFFu) << 16);
  g1[2] = (int)((dim0 >> 16) | ((dim1 & 0xFFFFu) << 16));
  g1[3] = (int)((dim1 >> 16) | (tile_x << 16));
  g1[4] = (int)tile_y;                                 // tile_dim1; tile_dim2=0
  g1[5] = (int)(unsigned int)stride_x;                 // tensor_dim0_stride[31:0]
  g1[6] = (int)(unsigned int)((stride_x >> 32) & 0xFFFFu);
  g1[7] = 0;
  i32x4 z4 = {0, 0, 0, 0};                             // groups 2/3: 2-D tensor
#if __clang_major__ >= 23
  i32x8 z8 = {0, 0, 0, 0, 0, 0, 0, 0};
  __builtin_amdgcn_tensor_load_to_lds(g0, g1, z4, z4, z8, 0);
#else
  __builtin_amdgcn_tensor_load_to_lds(g0, g1, z4, z4, 0);
#endif
}
#endif

// ---------------------------------------------------------------- convert
__global__ void __launch_bounds__(256) cvt_f32_f16(const float* __restrict__ in,
                                                   _Float16* __restrict__ out, int n4) {
  int i = blockIdx.x * 256 + threadIdx.x;
  if (i >= n4) return;
  float4 v = ((const float4*)in)[i];
  v4h o = { (_Float16)v.x, (_Float16)v.y, (_Float16)v.z, (_Float16)v.w };
  ((v4h*)out)[i] = o;
}

// ---------------------------------------------------------------- GEMM: C = A(f16) * W(f16)^T + bias
// MODE 0: Q -> [B,H,S,dk] f16   MODE 1: K -> [B,H,S,dk] f16
// MODE 2: V -> [B,H,dk,S] f16 (transposed)   MODE 3: out -> [M,N] f32
template <int MODE>
__global__ void __launch_bounds__(256) gemm16(const _Float16* __restrict__ A,
                                              const _Float16* __restrict__ W,
                                              const float* __restrict__ bias,
                                              void* __restrict__ outp) {
  constexpr int Kd = 1024, Nd = 1024;
  __shared__ _Float16 sA[2][128 * 40];
  __shared__ _Float16 sB[2][64 * 40];

  const int t = threadIdx.x;
  const int lane = t & 31, w = t >> 5;
  const int lr = lane & 15, hh = lane >> 4;
  const int waveM = w & 3, waveN = w >> 2;
  const int mBlock = blockIdx.y * 128;
  const int nBlock = blockIdx.x * 64;

  F8 acc[2][2];
  {
    float b0 = bias[nBlock + waveN * 32 + lr];
    float b1 = bias[nBlock + waveN * 32 + 16 + lr];
#pragma unroll
    for (int r = 0; r < 8; ++r) {
      acc[0][0].f[r] = b0; acc[1][0].f[r] = b0;
      acc[0][1].f[r] = b1; acc[1][1].f[r] = b1;
    }
  }

#if USE_TDM
  if (w == 0) {
    tdm_load_2d(&A[(size_t)mBlock * Kd], lds_off(&sA[0][0]), 32, 128, Kd, 3, 3);
    tdm_load_2d(&W[(size_t)nBlock * Kd], lds_off(&sB[0][0]), 32, 64,  Kd, 3, 3);
  }
#endif

  for (int kt = 0; kt < Kd / 32; ++kt) {
    const int k0 = kt * 32;
    const int cur = kt & 1;
    __syncthreads();                 // all waves done with buffer we now refill
#if USE_TDM
    if (w == 0) {
      if (kt + 1 < Kd / 32) {
        tdm_load_2d(&A[(size_t)mBlock * Kd + k0 + 32], lds_off(&sA[cur ^ 1][0]), 32, 128, Kd, 3, 3);
        tdm_load_2d(&W[(size_t)nBlock * Kd + k0 + 32], lds_off(&sB[cur ^ 1][0]), 32, 64,  Kd, 3, 3);
        __builtin_amdgcn_s_wait_tensorcnt(2);   // current tile's 2 ops complete
      } else {
        __builtin_amdgcn_s_wait_tensorcnt(0);
      }
    }
#else
#pragma unroll
    for (int i = 0; i < 2; ++i) {               // A tile: 128x32 halves
      int c = t + i * 256;
      int row = c >> 2, col = (c & 3) * 8;
      *(v8h*)&sA[cur][row * 40 + col] =
          *(const v8h*)&A[(size_t)(mBlock + row) * Kd + k0 + col];
    }
    {                                           // W tile: 64x32 halves
      int row = t >> 2, col = (t & 3) * 8;
      *(v8h*)&sB[cur][row * 40 + col] =
          *(const v8h*)&W[(size_t)(nBlock + row) * Kd + k0 + col];
    }
#endif
    __syncthreads();

    H16 af[2], bf[2];
#pragma unroll
    for (int mi = 0; mi < 2; ++mi) {
      const int row = waveM * 32 + mi * 16 + lr;
      af[mi].h[0] = *(const v8h*)&sA[cur][row * 40 + hh * 8];
      af[mi].h[1] = *(const v8h*)&sA[cur][row * 40 + 16 + hh * 8];
    }
#pragma unroll
    for (int ni = 0; ni < 2; ++ni) {
      const int row = waveN * 32 + ni * 16 + lr;
      bf[ni].h[0] = *(const v8h*)&sB[cur][row * 40 + hh * 8];
      bf[ni].h[1] = *(const v8h*)&sB[cur][row * 40 + 16 + hh * 8];
    }
#pragma unroll
    for (int mi = 0; mi < 2; ++mi)
#pragma unroll
      for (int ni = 0; ni < 2; ++ni)
        acc[mi][ni].v = __builtin_amdgcn_wmma_f32_16x16x32_f16(
            false, af[mi].v, false, bf[ni].v, (short)0, acc[mi][ni].v, false, false);
  }

#pragma unroll
  for (int mi = 0; mi < 2; ++mi) {
#pragma unroll
    for (int ni = 0; ni < 2; ++ni) {
      const int n = nBlock + waveN * 32 + ni * 16 + lr;
#pragma unroll
      for (int r = 0; r < 8; ++r) {
        const int m = mBlock + waveM * 32 + mi * 16 + hh * 8 + r;
        const float val = acc[mi][ni].f[r];
        if (MODE == 3) {
          ((float*)outp)[(size_t)m * Nd + n] = val;
        } else {
          const int b = m >> 11, s = m & 2047;
          const int head = n >> 6, d = n & 63;
          _Float16* o = (_Float16*)outp;
          if (MODE == 2)
            o[(((size_t)b * NHEAD + head) * DKH + d) * S_LEN + s] = (_Float16)val;
          else
            o[(((size_t)b * NHEAD + head) * S_LEN + s) * DKH + d] = (_Float16)val;
        }
      }
    }
  }
}

// ---------------------------------------------------------------- flash attention
// Q,K: [B,H,S,dk] f16 ; Vt: [B,H,dk,S] f16 ; out: [B,S,D] f16 (1/sqrt(dk) folded into Q)
__global__ void __launch_bounds__(128) attn_fwd(const _Float16* __restrict__ Q,
                                                const _Float16* __restrict__ Kk,
                                                const _Float16* __restrict__ Vt,
                                                _Float16* __restrict__ out) {
  __shared__ _Float16 sK[2][64 * 72];    // 64 keys x 64 d (padded)
  __shared__ _Float16 sV[2][64 * 72];    // 64 d x 64 keys (padded)
  __shared__ _Float16 sP[4 * 16 * 72];   // per-wave score/P transpose staging

  const int t = threadIdx.x;
  const int w = t >> 5, lane = t & 31;
  const int lr = lane & 15, hh = lane >> 4;
  const int bh = blockIdx.y;
  const int b = bh >> 4, head = bh & 15;
  const int q0 = blockIdx.x * 64 + w * 16;

  const _Float16* Qp = Q  + ((size_t)bh * S_LEN + q0) * DKH;
  const _Float16* Kp = Kk + (size_t)bh * S_LEN * DKH;
  const _Float16* Vp = Vt + (size_t)bh * DKH * S_LEN;

  H16 qf[2];
#pragma unroll
  for (int kd = 0; kd < 2; ++kd) {
    qf[kd].h[0] = *(const v8h*)&Qp[(size_t)lr * DKH + kd * 32 + hh * 8];
    qf[kd].h[1] = *(const v8h*)&Qp[(size_t)lr * DKH + kd * 32 + 16 + hh * 8];
#pragma unroll
    for (int e = 0; e < 16; ++e) qf[kd].e[e] = qf[kd].e[e] * (_Float16)0.125f;
  }

  F8 o[4];
#pragma unroll
  for (int di = 0; di < 4; ++di)
#pragma unroll
    for (int r = 0; r < 8; ++r) o[di].f[r] = 0.0f;
  float mrun = -1e30f, lrun = 0.0f;    // stats for row lr (dup in both halves)

#if USE_TDM
  if (w == 0) {
    tdm_load_2d(Kp, lds_off(&sK[0][0]), 64, 64, DKH,   4, 3);
    tdm_load_2d(Vp, lds_off(&sV[0][0]), 64, 64, S_LEN, 4, 3);
  }
#endif

  for (int it = 0; it < S_LEN / 64; ++it) {
    const int s0 = it * 64;
    const int cur = it & 1;
    __syncthreads();
#if USE_TDM
    if (w == 0) {
      if (s0 + 64 < S_LEN) {
        tdm_load_2d(&Kp[(size_t)(s0 + 64) * DKH], lds_off(&sK[cur ^ 1][0]), 64, 64, DKH,   4, 3);
        tdm_load_2d(&Vp[s0 + 64],                 lds_off(&sV[cur ^ 1][0]), 64, 64, S_LEN, 4, 3);
        __builtin_amdgcn_s_wait_tensorcnt(2);
      } else {
        __builtin_amdgcn_s_wait_tensorcnt(0);
      }
    }
#else
#pragma unroll
    for (int i = 0; i < 4; ++i) {                  // K tile 64x64, Vt tile 64x64
      int c = t + i * 128;
      int row = c >> 3, col = (c & 7) * 8;
      *(v8h*)&sK[cur][row * 72 + col] = *(const v8h*)&Kp[(size_t)(s0 + row) * DKH + col];
      *(v8h*)&sV[cur][row * 72 + col] = *(const v8h*)&Vp[(size_t)row * S_LEN + s0 + col];
    }
#endif
    __syncthreads();

    // ---- scores: Q(16x64) x K^T(64x64) -> 4 accumulators
    F8 sc[4];
#pragma unroll
    for (int ni = 0; ni < 4; ++ni) {
#pragma unroll
      for (int r = 0; r < 8; ++r) sc[ni].f[r] = 0.0f;
#pragma unroll
      for (int kd = 0; kd < 2; ++kd) {
        H16 bf;
        bf.h[0] = *(const v8h*)&sK[cur][(ni * 16 + lr) * 72 + kd * 32 + hh * 8];
        bf.h[1] = *(const v8h*)&sK[cur][(ni * 16 + lr) * 72 + kd * 32 + 16 + hh * 8];
        sc[ni].v = __builtin_amdgcn_wmma_f32_16x16x32_f16(
            false, qf[kd].v, false, bf.v, (short)0, sc[ni].v, false, false);
      }
    }

    // ---- transpose raw scores C-layout -> A-layout via per-wave LDS tile
    const int pbase = w * 16 * 72;
#pragma unroll
    for (int ni = 0; ni < 4; ++ni)
#pragma unroll
      for (int r = 0; r < 8; ++r)
        sP[pbase + (hh * 8 + r) * 72 + ni * 16 + lr] = (_Float16)sc[ni].f[r];
    __syncthreads();

    H16 ps[2];
#pragma unroll
    for (int kd = 0; kd < 2; ++kd) {
      ps[kd].h[0] = *(const v8h*)&sP[pbase + lr * 72 + kd * 32 + hh * 8];
      ps[kd].h[1] = *(const v8h*)&sP[pbase + lr * 72 + kd * 32 + 16 + hh * 8];
    }

    // ---- block row-max: in-lane over 32 elems, then merge lane halves
    _Float16 hmax = ps[0].e[0];
#pragma unroll
    for (int e = 1; e < 16; ++e) hmax = (ps[0].e[e] > hmax) ? ps[0].e[e] : hmax;
#pragma unroll
    for (int e = 0; e < 16; ++e) hmax = (ps[1].e[e] > hmax) ? ps[1].e[e] : hmax;
    float bm = (float)hmax;
    bm = fmaxf(bm, __shfl_xor(bm, 16));

    const float mnew = fmaxf(mrun, bm);
    const float corr = __expf(mrun - mnew);
    mrun = mnew;

    // ---- exp -> P fragments (already in A-layout) + row partial sum
    H16 pf[2];
    float rsum = 0.0f;
#pragma unroll
    for (int kd = 0; kd < 2; ++kd)
#pragma unroll
      for (int e = 0; e < 16; ++e) {
        float p = __expf((float)ps[kd].e[e] - mnew);
        rsum += p;
        pf[kd].e[e] = (_Float16)p;
      }
    rsum += __shfl_xor(rsum, 16);
    lrun = lrun * corr + rsum;

    // ---- rescale O accumulators (corr broadcast row-layout -> C-layout)
#pragma unroll
    for (int r = 0; r < 8; ++r) {
      float cc = __shfl(corr, hh * 8 + r);
      o[0].f[r] *= cc; o[1].f[r] *= cc; o[2].f[r] *= cc; o[3].f[r] *= cc;
    }

    // ---- O += P(16x64) x V(64x64)
#pragma unroll
    for (int di = 0; di < 4; ++di)
#pragma unroll
      for (int kd = 0; kd < 2; ++kd) {
        H16 bf;
        bf.h[0] = *(const v8h*)&sV[cur][(di * 16 + lr) * 72 + kd * 32 + hh * 8];
        bf.h[1] = *(const v8h*)&sV[cur][(di * 16 + lr) * 72 + kd * 32 + 16 + hh * 8];
        o[di].v = __builtin_amdgcn_wmma_f32_16x16x32_f16(
            false, pf[kd].v, false, bf.v, (short)0, o[di].v, false, false);
      }
  }

  const float linv = 1.0f / lrun;
#pragma unroll
  for (int r = 0; r < 8; ++r) {
    const float lc = __shfl(linv, hh * 8 + r);
    const int srow = q0 + hh * 8 + r;
#pragma unroll
    for (int di = 0; di < 4; ++di) {
      const int d = head * DKH + di * 16 + lr;
      out[((size_t)b * S_LEN + srow) * D_MODEL + d] = (_Float16)(o[di].f[r] * lc);
    }
  }
}

// ---------------------------------------------------------------- launch
extern "C" void kernel_launch(void* const* d_in, const int* in_sizes, int n_in,
                              void* d_out, int out_size, void* d_ws, size_t ws_size,
                              hipStream_t stream) {
  const float* x  = (const float*)d_in[0];
  const float* Wq = (const float*)d_in[1];
  const float* bq = (const float*)d_in[2];
  const float* Wk = (const float*)d_in[3];
  const float* bk = (const float*)d_in[4];
  const float* Wv = (const float*)d_in[5];
  const float* bv = (const float*)d_in[6];
  const float* Wo = (const float*)d_in[7];
  const float* bo = (const float*)d_in[8];

  constexpr size_t NX = (size_t)MTOT * D_MODEL;     // 8388608
  constexpr size_t NW = (size_t)D_MODEL * D_MODEL;  // 1048576
  _Float16* ws  = (_Float16*)d_ws;
  _Float16* xh  = ws;
  _Float16* wqh = xh  + NX;
  _Float16* wkh = wqh + NW;
  _Float16* wvh = wkh + NW;
  _Float16* woh = wvh + NW;
  _Float16* qws = woh + NW;
  _Float16* kws = qws + NX;
  _Float16* vws = kws + NX;
  _Float16* aws = vws + NX;

  cvt_f32_f16<<<dim3((unsigned)((NX / 4 + 255) / 256)), 256, 0, stream>>>(x, xh, (int)(NX / 4));
  cvt_f32_f16<<<dim3((unsigned)((NW / 4 + 255) / 256)), 256, 0, stream>>>(Wq, wqh, (int)(NW / 4));
  cvt_f32_f16<<<dim3((unsigned)((NW / 4 + 255) / 256)), 256, 0, stream>>>(Wk, wkh, (int)(NW / 4));
  cvt_f32_f16<<<dim3((unsigned)((NW / 4 + 255) / 256)), 256, 0, stream>>>(Wv, wvh, (int)(NW / 4));
  cvt_f32_f16<<<dim3((unsigned)((NW / 4 + 255) / 256)), 256, 0, stream>>>(Wo, woh, (int)(NW / 4));

  dim3 gg(D_MODEL / 64, MTOT / 128), gb(256);
  gemm16<0><<<gg, gb, 0, stream>>>(xh, wqh, bq, (void*)qws);
  gemm16<1><<<gg, gb, 0, stream>>>(xh, wkh, bk, (void*)kws);
  gemm16<2><<<gg, gb, 0, stream>>>(xh, wvh, bv, (void*)vws);

  attn_fwd<<<dim3(S_LEN / 64, BATCH * NHEAD), 128, 0, stream>>>(qws, kws, vws, aws);

  gemm16<3><<<gg, gb, 0, stream>>>(aws, woh, bo, d_out);
}
